// LocalGraphMessageBlock_65859028517423
// MI455X (gfx1250) — compile-verified, matrix operand order checked by reading, not run on hardware
//
#include <hip/hip_runtime.h>
#include <math.h>

// ---------------------------------------------------------------------------
// CDNA5 / gfx1250 implementation.
// All GEMMs use v_wmma_f32_16x16x32_bf16 (wave32, 16x16 C/D tiles, f32 acc).
// Weights are pre-transposed to bf16 K-contiguous layout so B fragments are
// straight 16B loads matching the ISA 16-bit B-matrix VGPR layout.
// Wave tile = 64 rows x 32 cols (4 A-frags from LDS, 2 B-frags from global,
// 8 accum tiles): each B fragment feeds 4 WMMAs. Within a block, every wave
// owns a distinct 32-column weight strip, so weight traffic from L2 is
// minimized (no intra-block redundancy; traffic scales as rows/64).
// ---------------------------------------------------------------------------

typedef __attribute__((ext_vector_type(16))) __bf16 v16bf;
typedef __attribute__((ext_vector_type(8)))  __bf16 v8bf;
typedef __attribute__((ext_vector_type(8)))  float  v8f;

union U16 { v16bf v; v8bf h[2]; };

// Load a 16-element bf16 fragment as two 16B chunks: per-lane layout is
// K = [base..base+7, base+16..base+23] with base = 0 (lanes<16) or 8 (lanes>=16).
__device__ __forceinline__ v16bf ld16u(const unsigned short* p) {
  U16 u;
  u.h[0] = *(const v8bf*)(p);
  u.h[1] = *(const v8bf*)(p + 16);
  return u.v;
}

__device__ __forceinline__ v8f wmma_bf16(v16bf a, v16bf b, v8f c) {
  // (neg_a, A, neg_b, B, c_mod, C, reuse_a, reuse_b)
  return __builtin_amdgcn_wmma_f32_16x16x32_bf16(false, a, false, b, (short)0, c,
                                                 false, false);
}

__device__ __forceinline__ unsigned short f2bf(float f) {  // round-to-nearest-even
  unsigned int u = __float_as_uint(f);
  u += 0x7fffu + ((u >> 16) & 1u);
  return (unsigned short)(u >> 16);
}

__device__ __forceinline__ float gelu_exact(float x) {
  return 0.5f * x * (1.0f + erff(x * 0.7071067811865475f));
}

__device__ __forceinline__ float wave_sum(float v) {
  #pragma unroll
  for (int s = 16; s > 0; s >>= 1) v += __shfl_xor(v, s, 32);
  return v;
}

// ---------------------------------------------------------------------------
// Weight prep: transpose + convert to bf16, K-contiguous per output column.
//   w1t[n][k] n<512 k<512 ; w2t[n][k] n<256 k<512
//   u1t[n][k] n<1024 k<256; u2t[n][k] n<256 k<1024
// ---------------------------------------------------------------------------
__global__ __launch_bounds__(256)
void k_prep(const float* __restrict__ w1, const float* __restrict__ w2,
            const float* __restrict__ u1, const float* __restrict__ u2,
            unsigned short* __restrict__ w1t, unsigned short* __restrict__ w2t,
            unsigned short* __restrict__ u1t, unsigned short* __restrict__ u2t) {
  int id = blockIdx.x * 256 + threadIdx.x;
  if (id < 262144) {                       // w1t: 512x512
    int n = id >> 9, k = id & 511;
    w1t[id] = f2bf(w1[(size_t)k * 512 + n]);
  } else if (id < 262144 + 131072) {       // w2t: 256x512
    int t = id - 262144;
    int n = t >> 9, k = t & 511;
    w2t[t] = f2bf(w2[(size_t)k * 256 + n]);
  } else if (id < 262144 + 131072 + 262144) {  // u1t: 1024x256
    int t = id - 262144 - 131072;
    int n = t >> 8, k = t & 255;
    u1t[t] = f2bf(u1[(size_t)k * 1024 + n]);
  } else if (id < 917504) {                // u2t: 256x1024
    int t = id - 262144 - 131072 - 262144;
    int n = t >> 10, k = t & 1023;
    u2t[t] = f2bf(u2[(size_t)k * 256 + n]);
  }
}

// ---------------------------------------------------------------------------
// LayerNorm over last dim (256). One wave per row. Optionally copies raw fp32
// input to xcopy (used to seed the message accumulator h2 = h).
// ---------------------------------------------------------------------------
__global__ __launch_bounds__(256)
void k_ln(const float* __restrict__ x, const float* __restrict__ g,
          const float* __restrict__ b, unsigned short* __restrict__ ybf,
          float* __restrict__ xcopy, int nrows) {
  int gw = (blockIdx.x * 256 + threadIdx.x) >> 5;
  int lane = threadIdx.x & 31;
  if (gw >= nrows) return;
  const float4* row = (const float4*)(x + (size_t)gw * 256);
  float4 a = row[lane];
  float4 c = row[lane + 32];
  float s = a.x + a.y + a.z + a.w + c.x + c.y + c.z + c.w;
  float q = a.x * a.x + a.y * a.y + a.z * a.z + a.w * a.w +
            c.x * c.x + c.y * c.y + c.z * c.z + c.w * c.w;
  s = wave_sum(s);
  q = wave_sum(q);
  float mu = s * (1.0f / 256.0f);
  float var = q * (1.0f / 256.0f) - mu * mu;
  float inv = rsqrtf(var + 1e-5f);
  int c0 = lane * 4, c1 = 128 + lane * 4;
  ushort4 o0, o1;
  o0.x = f2bf((a.x - mu) * inv * g[c0 + 0] + b[c0 + 0]);
  o0.y = f2bf((a.y - mu) * inv * g[c0 + 1] + b[c0 + 1]);
  o0.z = f2bf((a.z - mu) * inv * g[c0 + 2] + b[c0 + 2]);
  o0.w = f2bf((a.w - mu) * inv * g[c0 + 3] + b[c0 + 3]);
  o1.x = f2bf((c.x - mu) * inv * g[c1 + 0] + b[c1 + 0]);
  o1.y = f2bf((c.y - mu) * inv * g[c1 + 1] + b[c1 + 1]);
  o1.z = f2bf((c.z - mu) * inv * g[c1 + 2] + b[c1 + 2]);
  o1.w = f2bf((c.w - mu) * inv * g[c1 + 3] + b[c1 + 3]);
  ((ushort4*)(ybf + (size_t)gw * 256))[lane] = o0;
  ((ushort4*)(ybf + (size_t)gw * 256))[lane + 32] = o1;
  if (xcopy) {
    ((float4*)(xcopy + (size_t)gw * 256))[lane] = a;
    ((float4*)(xcopy + (size_t)gw * 256))[lane + 32] = c;
  }
}

// ---------------------------------------------------------------------------
// Fused message MLP for one (offset, direction) pass.
// Block = 256 threads = 8 waves; 64 edge rows of one graph n. Every wave
// covers all 64 rows and a distinct 32-column strip.
//   X[64x512] = concat(h_dst, h_src) staged in LDS (bf16)
//   hidden[64x512] = gelu(X@W1 + b1 + E@W1e)   (GEMM1, epilogue in LDS)
//   msg[64x256]    = hidden@W2 + b2            (GEMM2)
//   h2[n, c_dst, :] += msg * pair_valid        (unique rows per pass; passes
//                                               are stream-serialized)
// ---------------------------------------------------------------------------
__global__ __launch_bounds__(256)
void k_msg(const unsigned short* __restrict__ hin, const float* __restrict__ xyz,
           const int* __restrict__ valid,
           const unsigned short* __restrict__ w1t, const float* __restrict__ w1e,
           const float* __restrict__ b1,
           const unsigned short* __restrict__ w2t, const float* __restrict__ b2,
           float* __restrict__ h2, int off, int dir) {
  extern __shared__ __align__(16) unsigned char smem[];
  unsigned short* sX = (unsigned short*)smem;        // 64 x 528 bf16
  unsigned short* sH = sX + 64 * 528;                // 64 x 528 bf16
  float* sE = (float*)(sH + 64 * 528);               // 64 x 4
  float* sV = sE + 256;                              // 64

  const int cmax = 512 - off;
  const int n = blockIdx.x >> 3;
  const int m0 = (blockIdx.x & 7) * 64;
  const int tid = threadIdx.x;
  const int wid = tid >> 5, lane = tid & 31;
  const int l15 = lane & 15, khalf = (lane >> 4) * 8;

  // Stage X block (clamped rows for partial last block).
  #pragma unroll
  for (int it = 0; it < 16; ++it) {
    int ch = tid + it * 256;                 // 4096 chunks of 8 bf16
    int r = ch >> 6, k0 = (ch & 63) * 8;
    int m = m0 + r;
    m = m < cmax ? m : cmax - 1;
    int i = m, j = m + off;
    int ca = dir ? j : i, cb = dir ? i : j;  // row = [h_ca | h_cb]
    int cc = (k0 < 256) ? ca : cb;
    int kk = (k0 < 256) ? k0 : (k0 - 256);
    *(uint4*)(sX + r * 528 + k0) =
        *(const uint4*)(hin + ((size_t)(n * 512 + cc) * 256 + kk));
  }
  // Edge features + pair validity (row validity folded into sV).
  if (tid < 64) {
    int r = tid, m = m0 + r;
    bool inr = m < cmax;
    int mc = inr ? m : cmax - 1;
    const float* pi = xyz + ((size_t)n * 512 + mc) * 3;
    const float* pj = xyz + ((size_t)n * 512 + mc + off) * 3;
    float dx = pj[0] - pi[0], dy = pj[1] - pi[1], dz = pj[2] - pi[2];
    float d = fmaxf(sqrtf(dx * dx + dy * dy + dz * dz), 1e-6f);
    float sgn = dir ? -1.0f : 1.0f;
    float rid = 1.0f / d;
    sE[r * 4 + 0] = sgn * dx * rid;
    sE[r * 4 + 1] = sgn * dy * rid;
    sE[r * 4 + 2] = sgn * dz * rid;
    sE[r * 4 + 3] = d;
    sV[r] = (inr && valid[n * 512 + mc] && valid[n * 512 + mc + off]) ? 1.0f : 0.0f;
  }
  __syncthreads();

  const v8f vzero = {0.f, 0.f, 0.f, 0.f, 0.f, 0.f, 0.f, 0.f};

  // GEMM1: hidden = gelu(X @ W1 + b1 + E @ W1e). Wave: 64 rows x 32 cols x2.
  for (int cc = 0; cc < 2; ++cc) {
    const int colbase = wid * 32 + cc * 256;
    v8f acc[4][2];
    #pragma unroll
    for (int rr = 0; rr < 4; ++rr)
      #pragma unroll
      for (int t = 0; t < 2; ++t) acc[rr][t] = vzero;
    for (int kb = 0; kb < 512; kb += 32) {
      v16bf a[4];
      #pragma unroll
      for (int rr = 0; rr < 4; ++rr)
        a[rr] = ld16u(sX + (rr * 16 + l15) * 528 + kb + khalf);
      #pragma unroll
      for (int t = 0; t < 2; ++t) {
        v16bf b = ld16u(w1t + (size_t)(colbase + t * 16 + l15) * 512 + kb + khalf);
        #pragma unroll
        for (int rr = 0; rr < 4; ++rr)
          acc[rr][t] = wmma_bf16(a[rr], b, acc[rr][t]);
      }
    }
    #pragma unroll
    for (int t = 0; t < 2; ++t) {
      int col = colbase + t * 16 + l15;
      float e0 = w1e[col], e1 = w1e[512 + col], e2 = w1e[1024 + col],
            e3 = w1e[1536 + col];
      float bb = b1[col];
      #pragma unroll
      for (int rr = 0; rr < 4; ++rr) {
        #pragma unroll
        for (int r = 0; r < 8; ++r) {
          int row = rr * 16 + r + khalf;  // C/D layout: M = r + 8*(lane>=16)
          float xv = acc[rr][t][r] + bb + sE[row * 4 + 0] * e0 +
                     sE[row * 4 + 1] * e1 + sE[row * 4 + 2] * e2 +
                     sE[row * 4 + 3] * e3;
          sH[row * 528 + col] = f2bf(gelu_exact(xv));
        }
      }
    }
  }
  __syncthreads();

  // GEMM2: msg = hidden @ W2 + b2 ; accumulate into h2 (masked).
  {
    const int colbase = wid * 32;
    v8f acc[4][2];
    #pragma unroll
    for (int rr = 0; rr < 4; ++rr)
      #pragma unroll
      for (int t = 0; t < 2; ++t) acc[rr][t] = vzero;
    for (int kb = 0; kb < 512; kb += 32) {
      v16bf a[4];
      #pragma unroll
      for (int rr = 0; rr < 4; ++rr)
        a[rr] = ld16u(sH + (rr * 16 + l15) * 528 + kb + khalf);
      #pragma unroll
      for (int t = 0; t < 2; ++t) {
        v16bf b = ld16u(w2t + (size_t)(colbase + t * 16 + l15) * 512 + kb + khalf);
        #pragma unroll
        for (int rr = 0; rr < 4; ++rr)
          acc[rr][t] = wmma_bf16(a[rr], b, acc[rr][t]);
      }
    }
    #pragma unroll
    for (int t = 0; t < 2; ++t) {
      int col = colbase + t * 16 + l15;
      float bb = b2[col];
      #pragma unroll
      for (int rr = 0; rr < 4; ++rr) {
        #pragma unroll
        for (int r = 0; r < 8; ++r) {
          int row = rr * 16 + r + khalf;
          if (sV[row] != 0.0f) {
            int m = m0 + row;
            int cdst = dir ? (m + off) : m;
            size_t o = ((size_t)n * 512 + cdst) * 256 + col;
            h2[o] += acc[rr][t][r] + bb;
          }
        }
      }
    }
  }
}

// ---------------------------------------------------------------------------
// Fused update MLP: out = (h2 + gelu(hu@U1 + bu1)@U2 + bu2) * valid.
// Block handles 64 tokens; mid[64x1024] (bf16) lives in LDS.
// ---------------------------------------------------------------------------
__global__ __launch_bounds__(256)
void k_upd(const unsigned short* __restrict__ hu, const int* __restrict__ valid,
           const unsigned short* __restrict__ u1t, const float* __restrict__ bu1,
           const unsigned short* __restrict__ u2t, const float* __restrict__ bu2,
           const float* __restrict__ h2, float* __restrict__ out) {
  extern __shared__ __align__(16) unsigned char smem[];
  unsigned short* sA = (unsigned short*)smem;   // 64 x 272 bf16
  unsigned short* sM = sA + 64 * 272;           // 64 x 1040 bf16
  float* sVt = (float*)(sM + 64 * 1040);        // 64

  const int rows0 = blockIdx.x * 64;
  const int tid = threadIdx.x, wid = tid >> 5, lane = tid & 31;
  const int l15 = lane & 15, khalf = (lane >> 4) * 8;

  #pragma unroll
  for (int it = 0; it < 8; ++it) {
    int ch = tid + it * 256;
    int r = ch >> 5, k0 = (ch & 31) * 8;
    *(uint4*)(sA + r * 272 + k0) =
        *(const uint4*)(hu + ((size_t)(rows0 + r) * 256 + k0));
  }
  if (tid < 64) sVt[tid] = valid[rows0 + tid] ? 1.0f : 0.0f;
  __syncthreads();

  const v8f vzero = {0.f, 0.f, 0.f, 0.f, 0.f, 0.f, 0.f, 0.f};

  // GEMM1: mid = gelu(hu @ U1 + bu1), K=256, N=1024. Wave: 64 rows x 32 cols x4.
  for (int cc = 0; cc < 4; ++cc) {
    const int colbase = wid * 32 + cc * 256;
    v8f acc[4][2];
    #pragma unroll
    for (int rr = 0; rr < 4; ++rr)
      #pragma unroll
      for (int t = 0; t < 2; ++t) acc[rr][t] = vzero;
    for (int kb = 0; kb < 256; kb += 32) {
      v16bf a[4];
      #pragma unroll
      for (int rr = 0; rr < 4; ++rr)
        a[rr] = ld16u(sA + (rr * 16 + l15) * 272 + kb + khalf);
      #pragma unroll
      for (int t = 0; t < 2; ++t) {
        v16bf b = ld16u(u1t + (size_t)(colbase + t * 16 + l15) * 256 + kb + khalf);
        #pragma unroll
        for (int rr = 0; rr < 4; ++rr)
          acc[rr][t] = wmma_bf16(a[rr], b, acc[rr][t]);
      }
    }
    #pragma unroll
    for (int t = 0; t < 2; ++t) {
      int col = colbase + t * 16 + l15;
      float bb = bu1[col];
      #pragma unroll
      for (int rr = 0; rr < 4; ++rr) {
        #pragma unroll
        for (int r = 0; r < 8; ++r) {
          int row = rr * 16 + r + khalf;
          sM[row * 1040 + col] = f2bf(gelu_exact(acc[rr][t][r] + bb));
        }
      }
    }
  }
  __syncthreads();

  // GEMM2: out = (h2 + mid @ U2 + bu2) * valid, K=1024, N=256.
  {
    const int colbase = wid * 32;
    v8f acc[4][2];
    #pragma unroll
    for (int rr = 0; rr < 4; ++rr)
      #pragma unroll
      for (int t = 0; t < 2; ++t) acc[rr][t] = vzero;
    for (int kb = 0; kb < 1024; kb += 32) {
      v16bf a[4];
      #pragma unroll
      for (int rr = 0; rr < 4; ++rr)
        a[rr] = ld16u(sM + (rr * 16 + l15) * 1040 + kb + khalf);
      #pragma unroll
      for (int t = 0; t < 2; ++t) {
        v16bf b = ld16u(u2t + (size_t)(colbase + t * 16 + l15) * 1024 + kb + khalf);
        #pragma unroll
        for (int rr = 0; rr < 4; ++rr)
          acc[rr][t] = wmma_bf16(a[rr], b, acc[rr][t]);
      }
    }
    #pragma unroll
    for (int t = 0; t < 2; ++t) {
      int col = colbase + t * 16 + l15;
      float bb = bu2[col];
      #pragma unroll
      for (int rr = 0; rr < 4; ++rr) {
        #pragma unroll
        for (int r = 0; r < 8; ++r) {
          int row = rr * 16 + r + khalf;
          size_t o = (size_t)(rows0 + row) * 256 + col;
          out[o] = (h2[o] + acc[rr][t][r] + bb) * sVt[row];
        }
      }
    }
  }
}

// ---------------------------------------------------------------------------
extern "C" void kernel_launch(void* const* d_in, const int* in_sizes, int n_in,
                              void* d_out, int out_size, void* d_ws, size_t ws_size,
                              hipStream_t stream) {
  (void)in_sizes; (void)n_in; (void)out_size;
  const float* h     = (const float*)d_in[0];
  const float* xyz   = (const float*)d_in[1];
  const int*   valid = (const int*)d_in[2];
  const float* lnnw  = (const float*)d_in[3];
  const float* lnnb  = (const float*)d_in[4];
  const float* w1    = (const float*)d_in[5];
  const float* b1    = (const float*)d_in[6];
  const float* w2    = (const float*)d_in[7];
  const float* b2    = (const float*)d_in[8];
  const float* lnuw  = (const float*)d_in[9];
  const float* lnub  = (const float*)d_in[10];
  const float* u1    = (const float*)d_in[11];
  const float* bu1   = (const float*)d_in[12];
  const float* u2    = (const float*)d_in[13];
  const float* bu2   = (const float*)d_in[14];
  float* out = (float*)d_out;

  const size_t R = 256 * 512;  // tokens
  unsigned char* ws = (unsigned char*)d_ws;
  size_t p = 0;
  auto alloc = [&](size_t bytes) -> void* {
    void* r = ws + p;
    p += (bytes + 255) & ~(size_t)255;
    return r;
  };
  float*          h2  = (float*)alloc(R * 256 * 4);          // accumulator / residual
  unsigned short* hin = (unsigned short*)alloc(R * 256 * 2); // LN1(h), bf16
  unsigned short* hu  = (unsigned short*)alloc(R * 256 * 2); // LN2(h2), bf16
  unsigned short* w1t = (unsigned short*)alloc(512 * 512 * 2);
  unsigned short* w2t = (unsigned short*)alloc(256 * 512 * 2);
  unsigned short* u1t = (unsigned short*)alloc(1024 * 256 * 2);
  unsigned short* u2t = (unsigned short*)alloc(256 * 1024 * 2);
  if (p > ws_size) return;  // insufficient workspace

  hipLaunchKernelGGL(k_prep, dim3((917504 + 255) / 256), dim3(256), 0, stream,
                     w1, w2, u1, u2, w1t, w2t, u1t, u2t);

  // LN1 + seed h2 = h  (agg * valid_f == agg since pair-validity masks msgs)
  hipLaunchKernelGGL(k_ln, dim3(R / 8), dim3(256), 0, stream,
                     h, lnnw, lnnb, hin, h2, (int)R);

  const float* w1e = w1 + 512 * 512;  // rows 512..515 of w1 (edge-feature part)
  const size_t sm1 = (size_t)(64 * 528) * 2 * 2 + (256 + 64) * 4;
  const int offs[4] = {1, 2, 4, 8};
  for (int oi = 0; oi < 4; ++oi)
    for (int dir = 0; dir < 2; ++dir)
      hipLaunchKernelGGL(k_msg, dim3(256 * 8), dim3(256), sm1, stream,
                         hin, xyz, valid, w1t, w1e, b1, w2t, b2, h2,
                         offs[oi], dir);

  hipLaunchKernelGGL(k_ln, dim3(R / 8), dim3(256), 0, stream,
                     h2, lnuw, lnub, hu, (float*)nullptr, (int)R);

  const size_t sm2 = (size_t)(64 * 272 + 64 * 1040) * 2 + 64 * 4;
  hipLaunchKernelGGL(k_upd, dim3(R / 64), dim3(256), sm2, stream,
                     hu, valid, u1t, bu1, u2t, bu2, h2, out);
}
